// EncoderLayer_30485677867832
// MI455X (gfx1250) — matrix operand order repaired and checked by reading, not compiled, not run on hardware
//
#include <hip/hip_runtime.h>
#include <hip/hip_bf16.h>

// ---------------------------------------------------------------------------
// EncoderLayer on MI455X (gfx1250, wave32, WMMA).
// Exact algebraic simplification: reference applies softmax AFTER score@V, so
//   ctx = scale * Q @ (K^T @ V)   with per-(b,h) K^T V only 64x64.
// All heavy GEMMs: v_wmma_f32_16x16x32_bf16, f32 accumulation.
// B operands are kept transposed ([N,K]) so every fragment load is a
// contiguous 32B read; the main GEMM stages its B panel in LDS via
// global_load_async_to_lds_b128 (double buffered, s_wait_asynccnt).
// ---------------------------------------------------------------------------

#define USE_ASYNC_LDS 1

#define B_   2
#define L_   2048
#define DQ_  1024
#define H_   16
#define HD_  64
#define DFF_ 512
#define MTOK (B_ * L_)            // 4096 token rows

typedef __attribute__((ext_vector_type(16))) __bf16         v16bf;
typedef __attribute__((ext_vector_type(16))) unsigned short v16us;
typedef __attribute__((ext_vector_type(8)))  unsigned short v8us;
typedef __attribute__((ext_vector_type(8)))  float          v8f;

#define WMMA_BF16(a, b, c) \
  __builtin_amdgcn_wmma_f32_16x16x32_bf16(false, (a), false, (b), (short)0, (c), false, false)

__device__ __forceinline__ unsigned short f2bf(float f) {
  unsigned int u = __float_as_uint(f);
  u += 0x7FFFu + ((u >> 16) & 1u);            // round-to-nearest-even
  return (unsigned short)(u >> 16);
}

__device__ __forceinline__ float wave_sum(float v) {
#pragma unroll
  for (int o = 16; o > 0; o >>= 1) v += __shfl_xor(v, o, 32);
  return v;
}
__device__ __forceinline__ float wave_max(float v) {
#pragma unroll
  for (int o = 16; o > 0; o >>= 1) v = fmaxf(v, __shfl_xor(v, o, 32));
  return v;
}

__device__ __forceinline__ unsigned lds_addr32(const void* p) {
  // generic LDS pointer = {SHARED_BASE, lds_offset}; low 32 bits are the LDS address
  return (unsigned)(unsigned long long)(uintptr_t)p;
}

// --- A fragment (16x32 bf16, row-major source), contiguous K order:
//   lane half 0: elems 0..15 -> K 0..15
//   lane half 1: elems 0..15 -> K 16..31
__device__ __forceinline__ v16bf load_a_frag(const unsigned short* A, int lda,
                                             int row0, int k0, int lane) {
  int r = lane & 15, hs = (lane >> 4) & 1;
  const unsigned short* p = A + (long long)(row0 + r) * lda + k0 + hs * 16;
  v8us lo = *(const v8us*)p;          // K base .. +7
  v8us hi = *(const v8us*)(p + 8);    // K base+8 .. +15
  v16us u;
#pragma unroll
  for (int i = 0; i < 8; i++) { u[i] = lo[i]; u[i + 8] = hi[i]; }
  return __builtin_bit_cast(v16bf, u);
}

// --- B fragment from B^T ([N,K] row-major): contiguous 16 elems along K.
//   lane half 0: elem i -> K k0+i ; lane half 1: elem i -> K k0+16+i ; col = lane&15
__device__ __forceinline__ v16bf load_bt_frag(const unsigned short* BT, int ldt,
                                              int k0, int col0, int lane) {
  int c = lane & 15, hs = (lane >> 4) & 1;
  const unsigned short* p = BT + (long long)(col0 + c) * ldt + k0 + hs * 16;
  v16us u = *(const v16us*)p;
  return __builtin_bit_cast(v16bf, u);
}

// --- B fragment from an LDS panel laid out [128 cols][32 K] per buffer.
__device__ __forceinline__ v16bf lds_bt_frag(const unsigned short* panel,
                                             int col0, int lane) {
  int c = lane & 15, hs = (lane >> 4) & 1;
  const unsigned short* p = panel + (col0 + c) * 32 + hs * 16;
  v16us u = *(const v16us*)p;
  return __builtin_bit_cast(v16bf, u);
}

// --- C/D tile store. Per ISA: vgpr v holds row v + 8*(lane>=16), col = lane&15.
__device__ __forceinline__ void store_tile(float* Cf, unsigned short* Cbf, int ldc,
                                           int row0, int col0, v8f acc,
                                           const float* bias, float alpha, int relu,
                                           int storeT, int lane) {
  int c = lane & 15, hs = (lane >> 4) & 1;
  float bv = bias ? bias[col0 + c] : 0.0f;
#pragma unroll
  for (int v = 0; v < 8; v++) {
    int r = row0 + v + 8 * hs;
    float val = alpha * acc[v] + bv;
    if (relu) val = fmaxf(val, 0.0f);
    long long idx = storeT ? ((long long)(col0 + c) * ldc + r)
                           : ((long long)r * ldc + col0 + c);
    if (Cf)  Cf[idx]  = val;
    if (Cbf) Cbf[idx] = f2bf(val);
  }
}

// ---------------------------------------------------------------------------
// Main GEMM: C[M,N] = alpha * A[M,K] @ B[K,N] (+bias)(relu), B given as BT[N,K].
// 256 threads = 8 waves; wave tile 32x64 (8 WMMAs/k-step); block tile 128x128.
// B panel (128 x 32K, 8KB) staged in LDS via async copy, double buffered.
// All launch shapes are exact multiples of the block tile (no guards).
// ---------------------------------------------------------------------------
__global__ __launch_bounds__(256) void gemm_wmma_bf16(
    const unsigned short* __restrict__ A, const unsigned short* __restrict__ BT,
    float* Cf, unsigned short* Cbf, const float* bias,
    int M, int N, int K, int lda, int ldt, int ldc,
    float alpha, int relu, int storeT) {
  __shared__ unsigned short ldsB[2][128][32];
  int t = threadIdx.x;
  int lane = t & 31, wave = t >> 5;
  int wy = wave >> 1, wx = wave & 1;
  int bm = blockIdx.y * 128 + wy * 32;
  int bnBase = blockIdx.x * 128;
  int bn = bnBase + wx * 64;

  auto fill = [&](int buf, int k0) {
#pragma unroll
    for (int s = 0; s < 2; s++) {
      int seg = t + s * 256;                  // 512 x 16B segments = 8KB panel
      int row = seg >> 2, part = seg & 3;     // row: 0..127 (N), part: 0..3 (K/8)
      const unsigned short* g = BT + (long long)(bnBase + row) * ldt + k0 + part * 8;
      unsigned short* l = &ldsB[buf][row][part * 8];
#if USE_ASYNC_LDS
      unsigned la = lds_addr32(l);
      asm volatile("global_load_async_to_lds_b128 %0, %1, off"
                   :: "v"(la), "v"(g) : "memory");
#else
      *(v8us*)l = *(const v8us*)g;
#endif
    }
  };

  v8f acc0[4] = {}, acc1[4] = {};
  fill(0, 0);
  for (int k0 = 0, it = 0; k0 < K; k0 += 32, ++it) {
    int cur = it & 1;
    int more = (k0 + 32 < K);
    if (more) fill(cur ^ 1, k0 + 32);
#if USE_ASYNC_LDS
    if (more) asm volatile("s_wait_asynccnt 0x2" ::: "memory");
    else      asm volatile("s_wait_asynccnt 0x0" ::: "memory");
#endif
    __syncthreads();
    __builtin_prefetch(A + (long long)(bm + (lane & 15)) * lda + k0 + 64, 0, 1);
    v16bf fa0 = load_a_frag(A, lda, bm,      k0, lane);
    v16bf fa1 = load_a_frag(A, lda, bm + 16, k0, lane);
    const unsigned short* panel = &ldsB[cur][0][0];
#pragma unroll
    for (int j = 0; j < 4; j++) {
      v16bf fb = lds_bt_frag(panel, wx * 64 + j * 16, lane);
      acc0[j] = WMMA_BF16(fa0, fb, acc0[j]);
      acc1[j] = WMMA_BF16(fa1, fb, acc1[j]);
    }
    __syncthreads();
  }
#pragma unroll
  for (int j = 0; j < 4; j++) {
    store_tile(Cf, Cbf, ldc, bm,      bn + j * 16, acc0[j], bias, alpha, relu, storeT, lane);
    store_tile(Cf, Cbf, ldc, bm + 16, bn + j * 16, acc1[j], bias, alpha, relu, storeT, lane);
  }
}

// ---------------------------------------------------------------------------
// Mt[b,h] = (scale * K^T V)^T  (64x64, K-dim = L). Inputs are transposed
// projections Kt/Vt [DQ, MTOK], so both fragment loads are contiguous.
// One block per (b,h); 8 waves cover the 4x4 tile grid (2 tiles each).
// ---------------------------------------------------------------------------
__global__ __launch_bounds__(256) void ktv_wmma(
    const unsigned short* __restrict__ Kt, const unsigned short* __restrict__ Vt,
    unsigned short* __restrict__ Mt, float scale) {
  int z = blockIdx.x, b = z >> 4, h = z & 15;
  const unsigned short* Kb = Kt + (long long)(h * HD_) * MTOK + b * L_;
  const unsigned short* Vb = Vt + (long long)(h * HD_) * MTOK + b * L_;
  int lane = threadIdx.x & 31, wave = threadIdx.x >> 5;
  int ty = wave & 3, tx = (wave >> 2) * 2;
  v8f acc0 = {}, acc1 = {};
  for (int l0 = 0; l0 < L_; l0 += 32) {
    v16bf fa  = load_a_frag(Kb, MTOK, ty * 16, l0, lane);         // A = K^T
    v16bf fb0 = load_bt_frag(Vb, MTOK, l0, tx * 16, lane);        // B = V
    v16bf fb1 = load_bt_frag(Vb, MTOK, l0, (tx + 1) * 16, lane);
    acc0 = WMMA_BF16(fa, fb0, acc0);
    acc1 = WMMA_BF16(fa, fb1, acc1);
  }
  unsigned short* Mb = Mt + (long long)z * HD_ * HD_;
  store_tile(nullptr, Mb, HD_, ty * 16, tx * 16,       acc0, nullptr, scale, 0, 1, lane);
  store_tile(nullptr, Mb, HD_, ty * 16, (tx + 1) * 16, acc1, nullptr, scale, 0, 1, lane);
}

// ---------------------------------------------------------------------------
// ctx[b,l,h*64+d] = Q[b,l,h,:] @ M[b,h]  (N = K = 64). B from Mt (transposed).
// ---------------------------------------------------------------------------
__global__ __launch_bounds__(256) void qm_wmma(
    const unsigned short* __restrict__ Qh, const unsigned short* __restrict__ Mt,
    float* __restrict__ ctx) {
  int z = blockIdx.z, b = z >> 4, h = z & 15;
  const unsigned short* Qb = Qh + (long long)b * L_ * DQ_ + h * HD_;
  const unsigned short* Mb = Mt + (long long)z * HD_ * HD_;
  float* Cb = ctx + (long long)b * L_ * DQ_ + h * HD_;
  int lane = threadIdx.x & 31, wave = threadIdx.x >> 5;
  int bm = blockIdx.x * 128 + wave * 16;
  v8f acc[4] = {};
  for (int k0 = 0; k0 < HD_; k0 += 32) {
    v16bf fa = load_a_frag(Qb, DQ_, bm, k0, lane);
#pragma unroll
    for (int j = 0; j < 4; j++) {
      v16bf fb = load_bt_frag(Mb, HD_, k0, j * 16, lane);
      acc[j] = WMMA_BF16(fa, fb, acc[j]);
    }
  }
#pragma unroll
  for (int j = 0; j < 4; j++)
    store_tile(Cb, nullptr, DQ_, bm, j * 16, acc[j], nullptr, 1.0f, 0, 0, lane);
}

// ---------------------------------------------------------------------------
// Softmax over head dim (64) of ctx; emit bf16 in [B,L,H*HD] layout.
// ---------------------------------------------------------------------------
__global__ __launch_bounds__(256) void softmax64(
    const float* __restrict__ ctx, unsigned short* __restrict__ outh) {
  int row = blockIdx.x * 8 + (threadIdx.x >> 5);
  int lane = threadIdx.x & 31;
  const float* p = ctx + (long long)row * 64;
  float x0 = p[lane], x1 = p[lane + 32];
  float m = wave_max(fmaxf(x0, x1));
  float e0 = __expf(x0 - m), e1 = __expf(x1 - m);
  float s = wave_sum(e0 + e1);
  float inv = 1.0f / s;
  unsigned short* q = outh + (long long)row * 64;
  q[lane] = f2bf(e0 * inv);
  q[lane + 32] = f2bf(e1 * inv);
}

// ---------------------------------------------------------------------------
// out = LayerNorm(resid + y + addb) * g + beta ; optional f32 + bf16 outputs.
// ---------------------------------------------------------------------------
__global__ __launch_bounds__(256) void add_ln_kernel(
    const float* __restrict__ resid, const float* __restrict__ y,
    const float* __restrict__ addb, const float* __restrict__ g,
    const float* __restrict__ beta, float* outF, unsigned short* outBf) {
  const int n = DQ_;
  long long base = (long long)blockIdx.x * n;
  int t = threadIdx.x;
  float v[4], s = 0.0f, s2 = 0.0f;
#pragma unroll
  for (int i = 0; i < 4; i++) {
    int c = t + i * 256;
    float val = resid[base + c] + y[base + c] + (addb ? addb[c] : 0.0f);
    v[i] = val; s += val; s2 += val * val;
  }
  __shared__ float ls[8], ls2[8];
  s = wave_sum(s); s2 = wave_sum(s2);
  int lane = t & 31, w = t >> 5;
  if (lane == 0) { ls[w] = s; ls2[w] = s2; }
  __syncthreads();
  if (w == 0) {
    float a = (lane < 8) ? ls[lane] : 0.0f;
    float b = (lane < 8) ? ls2[lane] : 0.0f;
    a = wave_sum(a); b = wave_sum(b);
    if (lane == 0) { ls[0] = a; ls2[0] = b; }
  }
  __syncthreads();
  float mu = ls[0] / n;
  float var = ls2[0] / n - mu * mu;
  float inv = rsqrtf(var + 1e-5f);
#pragma unroll
  for (int i = 0; i < 4; i++) {
    int c = t + i * 256;
    float o = (v[i] - mu) * inv * g[c] + beta[c];
    if (outF)  outF[base + c] = o;
    if (outBf) outBf[base + c] = f2bf(o);
  }
}

// ---------------------------------------------------------------------------
__global__ __launch_bounds__(256) void cvt_f32_bf16(
    const float* __restrict__ src, unsigned short* __restrict__ dst, int n) {
  int i = blockIdx.x * 256 + threadIdx.x;
  if (i < n) dst[i] = f2bf(src[i]);
}

// dst[N,K] (bf16) = transpose(src[K,N] f32); LDS-tiled 32x32.
__global__ __launch_bounds__(256) void cvt_transpose_bf16(
    const float* __restrict__ src, unsigned short* __restrict__ dst,
    int rows, int cols) {
  __shared__ unsigned short tile[32][33];
  int bx = blockIdx.x * 32, by = blockIdx.y * 32;
  int tx = threadIdx.x & 31, ty = threadIdx.x >> 5;   // ty: 0..7
#pragma unroll
  for (int j = 0; j < 4; j++) {
    int r = by + ty + j * 8;
    tile[ty + j * 8][tx] = f2bf(src[(long long)r * cols + bx + tx]);
  }
  __syncthreads();
#pragma unroll
  for (int j = 0; j < 4; j++) {
    int rO = bx + ty + j * 8;
    dst[(long long)rO * rows + by + tx] = tile[tx][ty + j * 8];
  }
}

// ---------------------------------------------------------------------------
extern "C" void kernel_launch(void* const* d_in, const int* in_sizes, int n_in,
                              void* d_out, int out_size, void* d_ws, size_t ws_size,
                              hipStream_t stream) {
  const float* x    = (const float*)d_in[0];
  // d_in[1] = mask : intentionally unused (reference discards masked_fill)
  const float* Wq   = (const float*)d_in[2];
  const float* Wk   = (const float*)d_in[3];
  const float* Wv   = (const float*)d_in[4];
  const float* Wc   = (const float*)d_in[5];
  const float* bc   = (const float*)d_in[6];
  const float* ln1g = (const float*)d_in[7];
  const float* ln1b = (const float*)d_in[8];
  const float* W1   = (const float*)d_in[9];
  const float* b1   = (const float*)d_in[10];
  const float* W2   = (const float*)d_in[11];
  const float* b2   = (const float*)d_in[12];
  const float* ln2g = (const float*)d_in[13];
  const float* ln2b = (const float*)d_in[14];
  float* out = (float*)d_out;

  char* ws = (char*)d_ws;
  size_t off = 0;
  auto alloc = [&](size_t bytes) -> void* {
    off = (off + 255) & ~(size_t)255;
    void* p = ws + off;
    off += bytes;
    return p;
  };
  unsigned short* xh   = (unsigned short*)alloc((size_t)MTOK * DQ_ * 2);
  unsigned short* wqT  = (unsigned short*)alloc((size_t)DQ_ * DQ_ * 2);
  unsigned short* wkT  = (unsigned short*)alloc((size_t)DQ_ * DQ_ * 2);
  unsigned short* wvT  = (unsigned short*)alloc((size_t)DQ_ * DQ_ * 2);
  unsigned short* wcT  = (unsigned short*)alloc((size_t)DQ_ * DQ_ * 2);
  unsigned short* w1T  = (unsigned short*)alloc((size_t)DQ_ * DFF_ * 2);
  unsigned short* w2T  = (unsigned short*)alloc((size_t)DFF_ * DQ_ * 2);
  unsigned short* Qh   = (unsigned short*)alloc((size_t)MTOK * DQ_ * 2);
  unsigned short* Kt   = (unsigned short*)alloc((size_t)MTOK * DQ_ * 2);
  unsigned short* Vt   = (unsigned short*)alloc((size_t)MTOK * DQ_ * 2);
  unsigned short* Mt   = (unsigned short*)alloc((size_t)B_ * H_ * HD_ * HD_ * 2);
  float*          ctxf = (float*)alloc((size_t)MTOK * DQ_ * 4);
  unsigned short* ctxh = (unsigned short*)alloc((size_t)MTOK * DQ_ * 2);
  float*          proj = (float*)alloc((size_t)MTOK * DQ_ * 4);
  float*          attF = (float*)alloc((size_t)MTOK * DQ_ * 4);
  unsigned short* attH = (unsigned short*)alloc((size_t)MTOK * DQ_ * 2);
  unsigned short* h1h  = (unsigned short*)alloc((size_t)MTOK * DFF_ * 2);
  float*          ffnF = ctxf;   // reuse: ctxf dead after softmax64
  (void)ws_size; (void)in_sizes; (void)n_in; (void)out_size;

  dim3 blk(256);

  // 1) conversions: x -> bf16; weights -> transposed bf16 [N,K]
  cvt_f32_bf16<<<(MTOK * DQ_) / 256, blk, 0, stream>>>(x, xh, MTOK * DQ_);
  cvt_transpose_bf16<<<dim3(DQ_ / 32, DQ_ / 32), blk, 0, stream>>>(Wq, wqT, DQ_, DQ_);
  cvt_transpose_bf16<<<dim3(DQ_ / 32, DQ_ / 32), blk, 0, stream>>>(Wk, wkT, DQ_, DQ_);
  cvt_transpose_bf16<<<dim3(DQ_ / 32, DQ_ / 32), blk, 0, stream>>>(Wv, wvT, DQ_, DQ_);
  cvt_transpose_bf16<<<dim3(DQ_ / 32, DQ_ / 32), blk, 0, stream>>>(Wc, wcT, DQ_, DQ_);
  cvt_transpose_bf16<<<dim3(DFF_ / 32, DQ_ / 32), blk, 0, stream>>>(W1, w1T, DQ_, DFF_);
  cvt_transpose_bf16<<<dim3(DQ_ / 32, DFF_ / 32), blk, 0, stream>>>(W2, w2T, DFF_, DQ_);

  // 2) projections [4096,1024]@[1024,1024]; K/V stored transposed [1024,4096]
  dim3 gp(DQ_ / 128, MTOK / 128);
  gemm_wmma_bf16<<<gp, blk, 0, stream>>>(xh, wqT, nullptr, Qh, nullptr,
      MTOK, DQ_, DQ_, DQ_, DQ_, DQ_, 1.0f, 0, 0);
  gemm_wmma_bf16<<<gp, blk, 0, stream>>>(xh, wkT, nullptr, Kt, nullptr,
      MTOK, DQ_, DQ_, DQ_, DQ_, MTOK, 1.0f, 0, 1);
  gemm_wmma_bf16<<<gp, blk, 0, stream>>>(xh, wvT, nullptr, Vt, nullptr,
      MTOK, DQ_, DQ_, DQ_, DQ_, MTOK, 1.0f, 0, 1);

  // 3) per-head Mt = (scale * K^T V)^T   (scale = 1/sqrt(64))
  ktv_wmma<<<B_ * H_, blk, 0, stream>>>(Kt, Vt, Mt, 0.125f);

  // 4) ctx = Q @ M (f32, [B,L,H*HD])
  qm_wmma<<<dim3(L_ / 128, 1, B_ * H_), blk, 0, stream>>>(Qh, Mt, ctxf);

  // 5) softmax over head dim -> bf16
  softmax64<<<(B_ * L_ * H_) / 8, blk, 0, stream>>>(ctxf, ctxh);

  // 6) proj = ctx @ Wc (f32)
  gemm_wmma_bf16<<<gp, blk, 0, stream>>>(ctxh, wcT, proj, nullptr, nullptr,
      MTOK, DQ_, DQ_, DQ_, DQ_, DQ_, 1.0f, 0, 0);

  // 7) attn = LN(x + proj + bc) -> f32 + bf16
  add_ln_kernel<<<MTOK, blk, 0, stream>>>(x, proj, bc, ln1g, ln1b, attF, attH);

  // 8) h1 = relu(attn @ W1 + b1) -> bf16
  gemm_wmma_bf16<<<dim3(DFF_ / 128, MTOK / 128), blk, 0, stream>>>(
      attH, w1T, nullptr, h1h, b1, MTOK, DFF_, DQ_, DQ_, DQ_, DFF_, 1.0f, 1, 0);

  // 9) ffn = h1 @ W2 (f32)
  gemm_wmma_bf16<<<dim3(DQ_ / 128, MTOK / 128), blk, 0, stream>>>(
      h1h, w2T, ffnF, nullptr, nullptr, MTOK, DQ_, DFF_, DFF_, DFF_, DQ_, 1.0f, 0, 0);

  // 10) out = LN(attn + ffn + b2)
  add_ln_kernel<<<MTOK, blk, 0, stream>>>(attF, ffnF, b2, ln2g, ln2b, out, nullptr);
}
